// DNN_91225105367177
// MI455X (gfx1250) — matrix operand order repaired and checked by reading, not compiled, not run on hardware
//
#include <hip/hip_runtime.h>
#include <hip/hip_bf16.h>
#include <stdint.h>

// ---------------------------------------------------------------------------
// MI455X / gfx1250, wave32.  bf16 WMMA (v_wmma_f32_16x16x32_bf16) for the
// three GEMM stages; async global->LDS staging (ASYNCcnt) for the dominant
// pl1 GEMM; VALU for the tiny per-batch tail.
// ---------------------------------------------------------------------------

typedef __attribute__((ext_vector_type(16))) __bf16 v16bf;
typedef __attribute__((ext_vector_type(8)))  float  v8f;

union BFPack {
    uint4          q[2];   // 2 x 16B
    v16bf          v;      // 16 bf16
    unsigned short s[16];
};

static __device__ __forceinline__ unsigned short f2bf(float f) {
    union { float f; unsigned u; } in; in.f = f;
    unsigned u = in.u;
    unsigned r = u + 0x7FFFu + ((u >> 16) & 1u);   // round-to-nearest-even
    return (unsigned short)(r >> 16);
}

// Packed f32x2 -> bf16x2 (low | high<<16); native v_cvt_pk_bf16_f32 if present.
#if __has_builtin(__builtin_amdgcn_cvt_pk_bf16_f32)
static __device__ __forceinline__ unsigned int pack_bf16(float lo, float hi) {
    auto r = __builtin_amdgcn_cvt_pk_bf16_f32(lo, hi);
    unsigned int u;
    __builtin_memcpy(&u, &r, 4);
    return u;
}
#else
static __device__ __forceinline__ unsigned int pack_bf16(float lo, float hi) {
    return (unsigned int)f2bf(lo) | ((unsigned int)f2bf(hi) << 16);
}
#endif

static __device__ __forceinline__ v8f wmma_bf16(v16bf a, v16bf b, v8f c) {
    // D = A(16x32) * B(32x16) + C(16x16), f32 accumulate
    return __builtin_amdgcn_wmma_f32_16x16x32_bf16(
        /*neg_a*/false, a, /*neg_b*/false, b,
        /*c_mod*/(short)0, c, /*reuse_a*/false, /*reuse_b*/false);
}

// Async DMA: 16 bytes global -> LDS, tracked by ASYNCcnt (GV addressing mode).
static __device__ __forceinline__ void async_load_b128(const void* gptr,
                                                       uint32_t lds_off) {
    asm volatile("global_load_async_to_lds_b128 %0, %1, off"
                 :: "v"(lds_off), "v"((uint64_t)(uintptr_t)gptr)
                 : "memory");
}

// ---------------------------------------------------------------------------
// K0: f32 -> bf16 weight conversion (2 elements/thread, packed store)
// ---------------------------------------------------------------------------
__global__ void __launch_bounds__(256) k_cvt(const float* __restrict__ src,
                                             unsigned int* __restrict__ dst,
                                             int npairs) {
    int i = blockIdx.x * 256 + threadIdx.x;
    if (i < npairs) dst[i] = pack_bf16(src[2 * i], src[2 * i + 1]);
}

// ---------------------------------------------------------------------------
// K1: fused conv1(8->64) + conv2(64->64), both as WMMA with A = weights
// (M = out-channel), B = activations (N = position).  One wave handles a
// 16-position tile of one (b,l) row; conv1 K padded 8->32 with zeros.
// Output h2 is bf16 in layout [row=(b,l)][f = o*520 + pos]  (FLAT order).
// ---------------------------------------------------------------------------
__global__ void __launch_bounds__(128) k_conv(
    const float* __restrict__ x,              // (2048, 8, 520) f32
    const float* __restrict__ b1,             // 64
    const float* __restrict__ b2,             // 64
    const unsigned short* __restrict__ w1bf,  // 64x8  bf16
    const unsigned short* __restrict__ w2bf,  // 64x64 bf16
    unsigned short* __restrict__ h2)          // (2048, 33280) bf16
{
    __shared__ __align__(16) unsigned short lds_h1[4][16 * 64]; // 16pos x 64ch

    const int lane = threadIdx.x & 31;
    const int wave = threadIdx.x >> 5;
    const int tile = blockIdx.x * 4 + wave;         // 0 .. 2048*33-1
    const int row  = tile / 33;                     // (b*4 + l)
    const int pt   = tile % 33;                     // 16-position tile
    const int p0   = pt * 16;
    const int lp   = lane & 15;
    const int hi   = lane >> 4;                     // lane half
    const int pos  = p0 + lp;
    const bool pvalid = pos < 520;
    const int posc = pvalid ? pos : 519;

    // ---- A1: conv1 weights, 4 channel-tiles, K padded to 32 with zeros
    v16bf A1[4];
#pragma unroll
    for (int t = 0; t < 4; ++t) {
        BFPack pk;
        pk.q[0] = make_uint4(0, 0, 0, 0);
        pk.q[1] = make_uint4(0, 0, 0, 0);
        if (hi == 0)  // lanes 0-15 carry K=0..7 (real), rest zero
            pk.q[0] = *(const uint4*)(w1bf + (t * 16 + lp) * 8);
        A1[t] = pk.v;
    }
    // ---- A2: conv2 weights [4 ch-tiles][2 k-steps]
    v16bf A2[4][2];
#pragma unroll
    for (int t = 0; t < 4; ++t)
#pragma unroll
        for (int s = 0; s < 2; ++s) {
            const int ch = t * 16 + lp;
            const int kb = s * 32 + hi * 8;         // per-lane K base
            BFPack pk;
            pk.q[0] = *(const uint4*)(w2bf + ch * 64 + kb);
            pk.q[1] = *(const uint4*)(w2bf + ch * 64 + kb + 16);
            A2[t][s] = pk.v;
        }

    // ---- B1 from x: lane = position; lanes 16-31 (K=16..31) are the pad
    BFPack bx;
    bx.q[0] = make_uint4(0, 0, 0, 0);
    bx.q[1] = make_uint4(0, 0, 0, 0);
    if (hi == 0) {
        const float* xr = x + (size_t)row * 8 * 520 + posc;
        float xv[8];
#pragma unroll
        for (int c = 0; c < 8; ++c)
            xv[c] = pvalid ? xr[c * 520] : 0.0f;
        unsigned int* bq = (unsigned int*)&bx.q[0];
#pragma unroll
        for (int p = 0; p < 4; ++p)
            bq[p] = pack_bf16(xv[2 * p], xv[2 * p + 1]);
    }
    v16bf B1 = bx.v;

    // ---- conv1 WMMA + relu -> LDS [pos][ch] (one b128 store per tile)
    unsigned short* h1 = lds_h1[wave];
#pragma unroll
    for (int t = 0; t < 4; ++t) {
        v8f c;
        const float* bb = b1 + t * 16 + hi * 8;
#pragma unroll
        for (int j = 0; j < 8; ++j) c[j] = bb[j];
        v8f d = wmma_bf16(A1[t], B1, c);
        uint4 packed;
        unsigned int* pw = (unsigned int*)&packed;
#pragma unroll
        for (int p = 0; p < 4; ++p) {
            float v0 = d[2 * p]     > 0.0f ? d[2 * p]     : 0.0f;
            float v1 = d[2 * p + 1] > 0.0f ? d[2 * p + 1] : 0.0f;
            pw[p] = pack_bf16(v0, v1);
        }
        // ch = t*16 + hi*8 + j, j=0..7 contiguous, 16B aligned
        *(uint4*)(h1 + lp * 64 + t * 16 + hi * 8) = packed;
    }
    __syncthreads();

    // ---- B2 from LDS (lane = position, table-order K in channel dim)
    v16bf B2[2];
#pragma unroll
    for (int s = 0; s < 2; ++s) {
        const int ks = s * 32 + hi * 8;
        BFPack pk;
        pk.q[0] = *(const uint4*)(h1 + lp * 64 + ks);
        pk.q[1] = *(const uint4*)(h1 + lp * 64 + ks + 16);
        B2[s] = pk.v;
    }

    // ---- conv2 (K=64, two chained WMMAs) + relu -> h2 (channel-major FLAT)
    unsigned short* out = h2 + (size_t)row * 33280;
#pragma unroll
    for (int t = 0; t < 4; ++t) {
        v8f c;
        const float* bb = b2 + t * 16 + hi * 8;
#pragma unroll
        for (int j = 0; j < 8; ++j) c[j] = bb[j];
        v8f d = wmma_bf16(A2[t][0], B2[0], c);
        d     = wmma_bf16(A2[t][1], B2[1], d);
        if (pvalid) {
#pragma unroll
            for (int j = 0; j < 8; ++j) {
                float v = d[j] > 0.0f ? d[j] : 0.0f;
                int ch = t * 16 + hi * 8 + j;
                out[ch * 520 + pos] = f2bf(v);
            }
        }
    }
}

// ---------------------------------------------------------------------------
// K2: pl1 GEMM  act1 = relu( h2[2048 x 33280] * pl1_w^T[33280 x 256] + b )
// Block = 256 threads (8 waves), block tile = 32 rows x 64 cols, K-step = 64.
// A/B staged global->LDS with global_load_async_to_lds_b128 (double buffer,
// ASYNCcnt), WMMAs fed from LDS.  LDS rows padded to 80 elems (160B).
// ---------------------------------------------------------------------------
#define PL1_K   33280
#define LDSTR   80                     // padded row stride (elements)
#define A_ELEMS (32 * LDSTR)           // 2560 elements = 5120 B
#define SLOT_EL (A_ELEMS + 64 * LDSTR) // 7680 elements = 15360 B per slot

__global__ void __launch_bounds__(256) k_pl1(
    const unsigned short* __restrict__ h2,    // 2048 x 33280 bf16
    const unsigned short* __restrict__ wbf,   // 256 x 33280 bf16
    const float* __restrict__ bias,           // 256
    float* __restrict__ act1)                 // 2048 x 256 f32
{
    __shared__ __align__(16) unsigned short smem[2][SLOT_EL];  // 30720 B

    const int tid  = threadIdx.x;
    const int lane = tid & 31;
    const int wave = tid >> 5;
    const int rb = blockIdx.x >> 2;          // 0..63  (row block)
    const int cb = blockIdx.x & 3;           // 0..3   (col block)
    const int R0 = rb * 32, C0 = cb * 64;
    const int lp = lane & 15, hi = lane >> 4;

    const uint32_t ldsbase = (uint32_t)(uintptr_t)(&smem[0][0]);

    // --- per-thread staging chunks (16B each): 1 for A, 2 for B per K-step
    const int arow = tid >> 3;               // 0..31
    const int akc  = (tid & 7) * 8;          // element offset within K-step
    const uint64_t aG  = (uint64_t)(uintptr_t)(h2 + (size_t)(R0 + arow) * PL1_K + akc);
    const uint32_t aL  = (uint32_t)((arow * LDSTR + akc) * 2);

    const int bcol = tid >> 3;               // 0..31 (and +32 for 2nd chunk)
    const int bkc  = (tid & 7) * 8;
    const uint64_t bG0 = (uint64_t)(uintptr_t)(wbf + (size_t)(C0 + bcol) * PL1_K + bkc);
    const uint64_t bG1 = (uint64_t)(uintptr_t)(wbf + (size_t)(C0 + 32 + bcol) * PL1_K + bkc);
    const uint32_t bL0 = (uint32_t)((A_ELEMS + bcol * LDSTR + bkc) * 2);
    const uint32_t bL1 = (uint32_t)((A_ELEMS + (32 + bcol) * LDSTR + bkc) * 2);

    auto issue = [&](int it) {
        const uint64_t go = (uint64_t)it * 128;              // 64 elems * 2B
        const uint32_t lb = ldsbase + (uint32_t)(it & 1) * (SLOT_EL * 2);
        async_load_b128((const void*)(aG  + go), lb + aL);
        async_load_b128((const void*)(bG0 + go), lb + bL0);
        async_load_b128((const void*)(bG1 + go), lb + bL1);
    };

    // --- this wave's 16x16 output tile within the 32x64 block tile
    const int rw = (wave >> 2) * 16;         // 0 / 16
    const int cw = (wave & 3) * 16;          // 0 / 16 / 32 / 48

    v8f acc;
    const float bv = bias[C0 + cw + lp];     // C varies with N = lane&15
#pragma unroll
    for (int j = 0; j < 8; ++j) acc[j] = bv;

    issue(0);

    for (int it = 0; it < PL1_K / 64; ++it) {         // 520 iterations
        if (it + 1 < PL1_K / 64) {
            issue(it + 1);
            asm volatile("s_wait_asynccnt 0x3" ::: "memory");
        } else {
            asm volatile("s_wait_asynccnt 0x0" ::: "memory");
        }
        __syncthreads();

        const unsigned short* A  = smem[it & 1];
        const unsigned short* Bm = smem[it & 1] + A_ELEMS;
#pragma unroll
        for (int s = 0; s < 2; ++s) {
            BFPack a, b;
            const int ab = (rw + lp) * LDSTR + s * 32 + hi * 8;
            a.q[0] = *(const uint4*)(A + ab);
            a.q[1] = *(const uint4*)(A + ab + 16);
            const int bb = (cw + lp) * LDSTR + s * 32 + hi * 8;
            b.q[0] = *(const uint4*)(Bm + bb);
            b.q[1] = *(const uint4*)(Bm + bb + 16);
            acc = wmma_bf16(a.v, b.v, acc);
        }
        __syncthreads();   // protect slot before it+2's async writes land
    }

    float* orow = act1 + (size_t)(R0 + rw + hi * 8) * 256 + C0 + cw + lp;
#pragma unroll
    for (int j = 0; j < 8; ++j) {
        float v = acc[j] > 0.0f ? acc[j] : 0.0f;
        orow[j * 256] = v;
    }
}

// ---------------------------------------------------------------------------
// K3: pl2 (256->128 per row) + phase normalization -> Theta out, tr/ti ws
// ---------------------------------------------------------------------------
__global__ void __launch_bounds__(128) k_pl2(
    const float* __restrict__ act1,   // 2048 x 256
    const float* __restrict__ w,      // 128 x 256
    const float* __restrict__ bias,   // 128
    float* __restrict__ theta_out,    // 2048*64*2 (B,L,N,2)
    float* __restrict__ tr, float* __restrict__ ti)
{
    __shared__ float Rl[128];
    const int row = blockIdx.x;
    const int j = threadIdx.x;
    const float* a  = act1 + (size_t)row * 256;
    const float* wr = w + (size_t)j * 256;
    float s = bias[j];
    for (int i = 0; i < 256; ++i) s += a[i] * wr[i];
    Rl[j] = s;
    __syncthreads();
    if (j < 64) {
        float pr = Rl[j], pi = Rl[64 + j];
        float n = sqrtf(pr * pr + pi * pi);
        n = n > 1e-12f ? n : 1e-12f;
        float trv = pr / n, tiv = pi / n;
        theta_out[(row * 64 + j) * 2 + 0] = trv;
        theta_out[(row * 64 + j) * 2 + 1] = tiv;
        tr[row * 64 + j] = trv;
        ti[row * 64 + j] = tiv;
    }
}

// ---------------------------------------------------------------------------
// K4: per-batch tail: H einsums + beamformer MLPs + softmax + W scaling.
// One 128-thread workgroup per batch element.
// ---------------------------------------------------------------------------
__global__ void __launch_bounds__(128) k_tail(
    const float* __restrict__ Hre, const float* __restrict__ Him,   // (B,8,64,4,4)
    const float* __restrict__ chre, const float* __restrict__ chim, // (B,4,8)
    const float* __restrict__ b1w, const float* __restrict__ b1b,
    const float* __restrict__ b2w, const float* __restrict__ b2b,
    const float* __restrict__ b3w, const float* __restrict__ b3b,
    const float* __restrict__ p1w, const float* __restrict__ p1b,
    const float* __restrict__ p2w, const float* __restrict__ p2b,
    const float* __restrict__ tr, const float* __restrict__ ti,
    float* __restrict__ outW, float* __restrict__ outMu)
{
    __shared__ float str[256], sti[256];       // [l*64 + n]
    __shared__ float part[4][32][4];           // per-l partials per (k,m)
    __shared__ float cH[64];
    __shared__ float w0[128], q1[128], w1[128], w2[80];
    __shared__ float mu_s[2], scs;

    const int b = blockIdx.x;
    const int t = threadIdx.x;

    str[t] = tr[b * 256 + t];           str[t + 128] = tr[b * 256 + t + 128];
    sti[t] = ti[b * 256 + t];           sti[t + 128] = ti[b * 256 + t + 128];
    __syncthreads();

    // einsum partials: thread -> (km = t&31, l = t>>5)
    {
        const int km = t & 31, l = t >> 5;
        const int k = km >> 3, m = km & 7;
        const size_t base = ((size_t)b * 8 + m) * 1024 + l * 4 + k;
        const float* hre = Hre + base;
        const float* him = Him + base;
        float a = 0, bb = 0, c = 0, d = 0;
        for (int n = 0; n < 64; ++n) {
            float hr = hre[n * 16], hm = him[n * 16];
            float trv = str[l * 64 + n], tiv = sti[l * 64 + n];
            a += hr * trv;  bb += hr * tiv;
            c += hm * trv;  d += hm * tiv;
        }
        part[l][km][0] = a; part[l][km][1] = bb;
        part[l][km][2] = c; part[l][km][3] = d;
    }
    __syncthreads();

    if (t < 32) {
        const int k = t >> 3, m = t & 7;
        float a = 0, bb = 0, c = 0, d = 0;
        for (int l = 0; l < 4; ++l) {
            a += part[l][t][0]; bb += part[l][t][1];
            c += part[l][t][2]; d += part[l][t][3];
        }
        float top = a + d;        // Hre.tr + Him.ti
        float bot = bb - c;       // Hre.ti - Him.tr
        cH[k * 16 + m]     = top + chre[b * 32 + k * 8 + m];
        cH[k * 16 + 8 + m] = bot + chim[b * 32 + k * 8 + m];
    }
    __syncthreads();

    // b1 & p1 (64 -> 128) + relu
    {
        float s1 = b1b[t], s2 = p1b[t];
        const float* w1r = b1w + t * 64;
        const float* p1r = p1w + t * 64;
        for (int i = 0; i < 64; ++i) {
            float h = cH[i];
            s1 += w1r[i] * h;
            s2 += p1r[i] * h;
        }
        w0[t] = s1 > 0.0f ? s1 : 0.0f;
        q1[t] = s2 > 0.0f ? s2 : 0.0f;
    }
    __syncthreads();

    // b2 (128 -> 128) + relu
    {
        float s = b2b[t];
        const float* wr = b2w + t * 128;
        for (int i = 0; i < 128; ++i) s += wr[i] * w0[i];
        w1[t] = s > 0.0f ? s : 0.0f;
    }
    __syncthreads();

    // b3 (128 -> 80), p2 (128 -> 2)
    if (t < 80) {
        float s = b3b[t];
        const float* wr = b3w + t * 128;
        for (int i = 0; i < 128; ++i) s += wr[i] * w1[i];
        w2[t] = s;
    }
    if (t >= 96 && t < 98) {
        const int j = t - 96;
        float s = p2b[j];
        const float* wr = p2w + j * 128;
        for (int i = 0; i < 128; ++i) s += wr[i] * q1[i];
        mu_s[j] = s;
    }
    __syncthreads();

    if (t == 0) {
        float m0 = mu_s[0], m1 = mu_s[1];
        float mx = m0 > m1 ? m0 : m1;
        float e0 = expf(m0 - mx), e1 = expf(m1 - mx);
        float inv = 1.0f / (e0 + e1);
        mu_s[0] = e0 * inv;
        mu_s[1] = e1 * inv;
        float nn = 0.0f;
        for (int i = 0; i < 80; ++i) nn += w2[i] * w2[i];
        float norm = sqrtf(nn);
        norm = norm > 1e-12f ? norm : 1e-12f;
        scs = sqrtf(10.0f) * sqrtf(mu_s[0]) / norm;   // sqrt(PT)*sqrt(mu0)/||W||
    }
    __syncthreads();

    if (t < 80) outW[b * 80 + t] = w2[t] * scs;
    if (t < 2)  outMu[b * 2 + t] = mu_s[t];
}

// ---------------------------------------------------------------------------
// Host launcher
// ---------------------------------------------------------------------------
extern "C" void kernel_launch(void* const* d_in, const int* in_sizes, int n_in,
                              void* d_out, int out_size, void* d_ws, size_t ws_size,
                              hipStream_t stream)
{
    const float* x      = (const float*)d_in[0];
    const float* Hre    = (const float*)d_in[1];
    const float* Him    = (const float*)d_in[2];
    const float* chre   = (const float*)d_in[3];
    const float* chim   = (const float*)d_in[4];
    const float* conv1w = (const float*)d_in[5];
    const float* conv1b = (const float*)d_in[6];
    const float* conv2w = (const float*)d_in[7];
    const float* conv2b = (const float*)d_in[8];
    const float* pl1w   = (const float*)d_in[9];
    const float* pl1b   = (const float*)d_in[10];
    const float* pl2w   = (const float*)d_in[11];
    const float* pl2b   = (const float*)d_in[12];
    const float* b1w = (const float*)d_in[13]; const float* b1b = (const float*)d_in[14];
    const float* b2w = (const float*)d_in[15]; const float* b2b = (const float*)d_in[16];
    const float* b3w = (const float*)d_in[17]; const float* b3b = (const float*)d_in[18];
    const float* p1w = (const float*)d_in[19]; const float* p1b = (const float*)d_in[20];
    const float* p2w = (const float*)d_in[21]; const float* p2b = (const float*)d_in[22];

    // workspace carve-out (~156 MB), 256B aligned segments
    char* ws = (char*)d_ws;
    size_t off = 0;
    auto alloc = [&](size_t bytes) -> void* {
        void* p = ws + off;
        off = (off + bytes + 255) & ~(size_t)255;
        return p;
    };
    unsigned short* w1bf  = (unsigned short*)alloc((size_t)512 * 2);
    unsigned short* w2bf  = (unsigned short*)alloc((size_t)4096 * 2);
    unsigned short* pl1bf = (unsigned short*)alloc((size_t)8519680 * 2);
    unsigned short* h2    = (unsigned short*)alloc((size_t)2048 * 33280 * 2);
    float* act1 = (float*)alloc((size_t)2048 * 256 * 4);
    float* trb  = (float*)alloc((size_t)2048 * 64 * 4);
    float* tib  = (float*)alloc((size_t)2048 * 64 * 4);

    // all weight element counts are even -> pairwise packed conversion
    k_cvt<<<(256 + 255) / 256, 256, 0, stream>>>(conv1w, (unsigned int*)w1bf, 256);
    k_cvt<<<(2048 + 255) / 256, 256, 0, stream>>>(conv2w, (unsigned int*)w2bf, 2048);
    k_cvt<<<(4259840 + 255) / 256, 256, 0, stream>>>(pl1w, (unsigned int*)pl1bf, 4259840);

    // 2048 rows * 33 position-tiles = 67584 wave-tiles, 4 waves per block
    k_conv<<<16896, 128, 0, stream>>>(x, conv1b, conv2b, w1bf, w2bf, h2);

    // 64 row-blocks * 4 col-blocks, 256 threads (8 waves) each
    k_pl1<<<256, 256, 0, stream>>>(h2, pl1bf, pl1b, act1);

    float* out = (float*)d_out;
    float* thetaOut = out + 40960;            // after W (512*80)
    float* muOut    = out + 40960 + 262144;   // after Theta (512*4*64*2)

    k_pl2<<<2048, 128, 0, stream>>>(act1, pl2w, pl2b, thetaOut, trb, tib);
    k_tail<<<512, 128, 0, stream>>>(Hre, Him, chre, chim,
                                    b1w, b1b, b2w, b2b, b3w, b3b,
                                    p1w, p1b, p2w, p2b,
                                    trb, tib, out, muOut);
}
